// qODEDynamics_58342835749471
// MI455X (gfx1250) — matrix operand order repaired and checked by reading, not compiled, not run on hardware
//
#include <hip/hip_runtime.h>
#include <hip/hip_bf16.h>

typedef __attribute__((ext_vector_type(16))) _Float16 v16h;
typedef __attribute__((ext_vector_type(8)))  _Float16 v8h;
typedef __attribute__((ext_vector_type(8)))  float    v8f;

#define NN 1024
#define LL 32
#define HH 64
#define JC 8              // j-chunks
#define JCHUNK (NN / JC)  // 128 j per chunk
#define IWAVES 8          // i-rows (waves) per block

#if __has_builtin(__builtin_amdgcn_tanhf)
__device__ __forceinline__ float fast_tanh(float x) {
  return __builtin_amdgcn_tanhf(x);
}
#else
__device__ __forceinline__ float fast_tanh(float x) {
  float cx = fminf(fmaxf(x, -9.0f), 9.0f);
  float e = __expf(2.0f * cx);
  return (e - 1.0f) * __builtin_amdgcn_rcpf(e + 1.0f);
}
#endif
__device__ __forceinline__ float fast_sig(float x) {
  return __builtin_amdgcn_rcpf(1.0f + __expf(-x));
}

// ---------------------------------------------------------------------------
// Kernel A: per-node precompute. One 64-thread block per node.
// self_term (N x 32), a (N x 64), b (N x 64), hcA/hcB (N x 32), aggS (N)
// ---------------------------------------------------------------------------
__global__ __launch_bounds__(64) void node_kernel(
    const float* __restrict__ state,
    const float* __restrict__ Ws1, const float* __restrict__ bs1,
    const float* __restrict__ Ws2, const float* __restrict__ bs2,
    const float* __restrict__ Ws3, const float* __restrict__ bs3,
    const float* __restrict__ Wi1, const float* __restrict__ bi1,
    const float* __restrict__ Wc,  const float* __restrict__ aggr,
    float* __restrict__ self_t, float* __restrict__ av, float* __restrict__ bv,
    float* __restrict__ hcA, float* __restrict__ hcB, float* __restrict__ aggS) {
  int i = blockIdx.x;
  int t = threadIdx.x;  // 0..63
  __shared__ float hloc[LL];
  __shared__ float t1[HH];
  __shared__ float t2[HH];
  __shared__ float dh[LL];

  if (t < LL) hloc[t] = state[i * (LL + 1) + t];
  __syncthreads();

  float s1 = bs1[t];
#pragma unroll
  for (int d = 0; d < LL; ++d) s1 += hloc[d] * Ws1[t * LL + d];
  t1[t] = fast_tanh(s1);
  __syncthreads();

  float s2 = bs2[t];
#pragma unroll
  for (int k = 0; k < HH; ++k) s2 += t1[k] * Ws2[t * HH + k];
  t2[t] = fast_tanh(s2);
  __syncthreads();

  if (t < LL) {
    float s3 = bs3[t];
#pragma unroll
    for (int k = 0; k < HH; ++k) s3 += t2[k] * Ws3[t * HH + k];
    dh[t] = s3;
    self_t[i * LL + t] = s3;
  }
  __syncthreads();

  // a_i[t] = h.Wi1[t, 0:L] + dh.Wi1[t, 2L:3L]
  // b_i[t] = h.Wi1[t, L:2L] + dh.Wi1[t, 3L:4L] + bi1[t]
  float sa = 0.0f, sb = bi1[t];
#pragma unroll
  for (int d = 0; d < LL; ++d) {
    float hd = hloc[d], dhd = dh[d];
    sa += hd * Wi1[t * (4 * LL) + d]      + dhd * Wi1[t * (4 * LL) + 2 * LL + d];
    sb += hd * Wi1[t * (4 * LL) + LL + d] + dhd * Wi1[t * (4 * LL) + 3 * LL + d];
  }
  av[i * HH + t] = sa;
  bv[i * HH + t] = sb;

  if (t < LL) {
    float ca = 0.0f, cb = 0.0f;
#pragma unroll
    for (int d = 0; d < LL; ++d) {
      ca += hloc[d] * Wc[t * (2 * LL) + d];
      cb += hloc[d] * Wc[t * (2 * LL) + LL + d];
    }
    hcA[i * LL + t] = ca;
    hcB[i * LL + t] = cb;
  }
  if (t == 0) aggS[i] = fast_sig(aggr[i]);
}

// ---------------------------------------------------------------------------
// Kernel B: pairwise interaction via WMMA.
// Block: 8 waves (256 threads), wave w owns i = blockIdx.x*8 + w.
// Grid: (N/8, JC). Each block accumulates isum partials over its j-chunk.
// GEMM1 weights live in register B-fragments (used 8x/iter); GEMM2 weights
// are re-read from LDS (keeps allocation under the 256-VGPR direct window).
// j-tile staging is double-buffered: 1 barrier/iter, loads overlap compute.
// ---------------------------------------------------------------------------
__global__ __launch_bounds__(IWAVES * 32) void pair_kernel(
    const float* __restrict__ state,
    const float* __restrict__ Wi2, const float* __restrict__ bi2,
    const float* __restrict__ Wi3, const float* __restrict__ bi3,
    const float* __restrict__ phase_w, const float* __restrict__ bc,
    const float* __restrict__ av, const float* __restrict__ bv,
    const float* __restrict__ hcA, const float* __restrict__ hcB,
    const float* __restrict__ aggS, float* __restrict__ part) {
  const int tid  = threadIdx.x;
  const int wave = tid >> 5;
  const int lane = tid & 31;
  const int ln   = lane & 15;
  const int hi16 = lane >> 4;  // 0 or 1

  const int i  = blockIdx.x * IWAVES + wave;
  const int jc = blockIdx.y;

  // ---- LDS ----
  __shared__ __align__(16) float sh_hj[2][16 * LL];    // h[j][l], double-buffered
  __shared__ float sh_phj[2][16];                      // phase_j
  __shared__ __align__(16) float sh_hcBj[2][16 * LL];  // hcB[j][l]
  __shared__ __align__(16) float sh_bj[2][16 * HH];    // b_j vectors
  __shared__ __align__(16) float sh_hi[IWAVES * LL];   // h[i][l] for block's i rows
  __shared__ float sh_hcAi[IWAVES * LL];
  __shared__ __align__(32) float sh_ai[IWAVES * HH];   // a_i vectors
  __shared__ __align__(32) _Float16 sh_x2[IWAVES * 16 * HH];  // per-wave x2 scratch
  __shared__ __align__(32) _Float16 sh_w2f[4 * 2 * 32 * 16];  // Wi2^T B-fragments
  __shared__ __align__(32) _Float16 sh_w3f[2 * 2 * 32 * 16];  // Wi3^T B-fragments

  // ---- one-time staging: i-row data + weight B-fragments ----
  for (int idx = tid; idx < IWAVES * LL; idx += IWAVES * 32) {
    int w = idx >> 5, d = idx & 31;
    int ii = blockIdx.x * IWAVES + w;
    sh_hi[idx]   = state[ii * (LL + 1) + d];
    sh_hcAi[idx] = hcA[ii * LL + d];
  }
  for (int idx = tid; idx < IWAVES * HH; idx += IWAVES * 32) {
    int w = idx >> 6, k = idx & 63;
    sh_ai[idx] = av[(blockIdx.x * IWAVES + w) * HH + k];
  }
  // B fragment layout (16x16x32 f16 WMMA): lane l<16 -> col n = l, K = {0..7,16..23};
  // lane l>=16 -> col n = l-16, K = {8..15,24..31} (matches A table order).
  for (int idx = tid; idx < 4 * 2 * 32 * 16; idx += IWAVES * 32) {
    int f = idx >> 9;            // fragment id = t*2+s
    int l = (idx >> 4) & 31;     // lane
    int p = idx & 15;
    int t = f >> 1, s = f & 1;
    int n = t * 16 + (l & 15);
    int k = s * 32 + ((p >> 3) << 4) + ((l >> 4) << 3) + (p & 7);
    sh_w2f[idx] = (_Float16)Wi2[n * HH + k];  // Wi2^T[k][n]
  }
  for (int idx = tid; idx < 2 * 2 * 32 * 16; idx += IWAVES * 32) {
    int f = idx >> 9;
    int l = (idx >> 4) & 31;
    int p = idx & 15;
    int t = f >> 1, s = f & 1;
    int n = t * 16 + (l & 15);
    int k = s * 32 + ((p >> 3) << 4) + ((l >> 4) << 3) + (p & 7);
    sh_w3f[idx] = (_Float16)Wi3[n * HH + k];  // Wi3^T[k][n]
  }
  __syncthreads();

  // ---- hoist GEMM1 weight B-fragments into registers (8 uses / iter) ----
  v16h w2r[4][2];
#pragma unroll
  for (int t = 0; t < 4; ++t) {
#pragma unroll
    for (int s = 0; s < 2; ++s)
      w2r[t][s] = *(const v16h*)&sh_w2f[((t * 2 + s) * 32 + lane) * 16];
  }

  // ---- per-wave loop invariants ----
  const float phase_i = state[i * (LL + 1) + LL];
  const float aggi    = aggS[i];
  float b2l[4], b3l[2];
#pragma unroll
  for (int t = 0; t < 4; ++t) b2l[t] = bi2[t * 16 + ln];
#pragma unroll
  for (int t = 0; t < 2; ++t) b3l[t] = bi3[t * 16 + ln];
  const float pw0 = phase_w[ln],      pw1 = phase_w[16 + ln];
  const float bc0 = bc[ln],           bc1 = bc[16 + ln];
  const float hil0 = sh_hi[wave * LL + ln],   hil1 = sh_hi[wave * LL + 16 + ln];
  const float hca0 = sh_hcAi[wave * LL + ln], hca1 = sh_hcAi[wave * LL + 16 + ln];

  float sum0 = 0.0f, sum1 = 0.0f;

  auto stage = [&](int buf, int j0) {
    for (int idx = tid; idx < 16 * LL; idx += IWAVES * 32) {
      int jl = idx >> 5, d = idx & 31;
      sh_hj[buf][idx]   = state[(j0 + jl) * (LL + 1) + d];
      sh_hcBj[buf][idx] = hcB[(j0 + jl) * LL + d];
    }
    for (int idx = tid; idx < 16 * HH; idx += IWAVES * 32) {
      sh_bj[buf][idx] = bv[(j0 + (idx >> 6)) * HH + (idx & 63)];
    }
    if (tid < 16) sh_phj[buf][tid] = state[(j0 + tid) * (LL + 1) + LL];
  };

  const int TILES = JCHUNK / 16;
  stage(0, jc * JCHUNK);

  for (int jt = 0; jt < TILES; ++jt) {
    __syncthreads();  // staged data for this tile ready; prev tile fully consumed
    if (jt + 1 < TILES) stage((jt + 1) & 1, jc * JCHUNK + (jt + 1) * 16);
    const int buf = jt & 1;
    const int j0  = jc * JCHUNK + jt * 16;

    // ---- x1 A-fragments: M = j_local, K = hidden (64, 2 K-steps of 32) ----
    // A 16x32 f16 layout: lane-half 0 holds K blocks {0..7,16..23}, half 1 {8..15,24..31}
    v16h A1[2];
#pragma unroll
    for (int s = 0; s < 2; ++s) {
#pragma unroll
      for (int g = 0; g < 2; ++g) {
        const int kb = s * 32 + g * 16 + hi16 * 8;  // 8-contiguous K run, 32B aligned
        const float4* pa = (const float4*)&sh_ai[wave * HH + kb];
        const float4* pb = (const float4*)&sh_bj[buf][ln * HH + kb];
        float4 a0 = pa[0], a1 = pa[1];
        float4 b0 = pb[0], b1 = pb[1];
        A1[s][g * 8 + 0] = (_Float16)fast_tanh(a0.x + b0.x);
        A1[s][g * 8 + 1] = (_Float16)fast_tanh(a0.y + b0.y);
        A1[s][g * 8 + 2] = (_Float16)fast_tanh(a0.z + b0.z);
        A1[s][g * 8 + 3] = (_Float16)fast_tanh(a0.w + b0.w);
        A1[s][g * 8 + 4] = (_Float16)fast_tanh(a1.x + b1.x);
        A1[s][g * 8 + 5] = (_Float16)fast_tanh(a1.y + b1.y);
        A1[s][g * 8 + 6] = (_Float16)fast_tanh(a1.z + b1.z);
        A1[s][g * 8 + 7] = (_Float16)fast_tanh(a1.w + b1.w);
      }
    }

    // ---- GEMM1: x2_acc(16x64) = x1(16x64) @ Wi2^T(64x64) + bi2 ----
    v8f acc[4];
#pragma unroll
    for (int t = 0; t < 4; ++t) {
#pragma unroll
      for (int r = 0; r < 8; ++r) acc[t][r] = b2l[t];
    }
#pragma unroll
    for (int t = 0; t < 4; ++t) {
      acc[t] = __builtin_amdgcn_wmma_f32_16x16x32_f16(false, A1[0], false, w2r[t][0],
                                                      (short)0, acc[t], false, false);
      acc[t] = __builtin_amdgcn_wmma_f32_16x16x32_f16(false, A1[1], false, w2r[t][1],
                                                      (short)0, acc[t], false, false);
    }

    // ---- dfac(i, j0+ln): independent work to overlap WMMA latency ----
    float d2 = 0.0f;
#pragma unroll
    for (int d = 0; d < LL; ++d) {
      float df = sh_hi[wave * LL + d] - sh_hj[buf][ln * LL + d];
      d2 += df * df;
    }
    float dist = sqrtf(d2) + 1e-6f;
    float dfac = fminf(__builtin_amdgcn_rcpf(dist), 2.0f);

    // ---- tanh + transpose round-trip through LDS (D-layout -> A-layout) ----
    _Float16* myx2 = &sh_x2[wave * 16 * HH];
#pragma unroll
    for (int t = 0; t < 4; ++t) {
#pragma unroll
      for (int r = 0; r < 8; ++r) {
        int m = r + hi16 * 8;
        int n = t * 16 + ln;
        myx2[m * HH + n] = (_Float16)fast_tanh(acc[t][r]);
      }
    }
    v16h A2[2];
#pragma unroll
    for (int s = 0; s < 2; ++s) {
      const _Float16* base = &myx2[ln * HH + s * 32 + hi16 * 8];
      v8h lo = *(const v8h*)base;
      v8h hi = *(const v8h*)(base + 16);
#pragma unroll
      for (int p = 0; p < 8; ++p) { A2[s][p] = lo[p]; A2[s][p + 8] = hi[p]; }
    }

    // ---- GEMM2: k(16x32) = x2(16x64) @ Wi3^T(64x32) + bi3 ----
    v8f kacc[2];
#pragma unroll
    for (int t = 0; t < 2; ++t) {
#pragma unroll
      for (int r = 0; r < 8; ++r) kacc[t][r] = b3l[t];
    }
#pragma unroll
    for (int t = 0; t < 2; ++t) {
      v16h b0 = *(const v16h*)&sh_w3f[((t * 2 + 0) * 32 + lane) * 16];
      v16h b1 = *(const v16h*)&sh_w3f[((t * 2 + 1) * 32 + lane) * 16];
      kacc[t] = __builtin_amdgcn_wmma_f32_16x16x32_f16(false, A2[0], false, b0,
                                                       (short)0, kacc[t], false, false);
      kacc[t] = __builtin_amdgcn_wmma_f32_16x16x32_f16(false, A2[1], false, b1,
                                                       (short)0, kacc[t], false, false);
    }

    // ---- elementwise combine on C/D layout ----
#pragma unroll
    for (int r = 0; r < 8; ++r) {
      int m = r + hi16 * 8;
      int j = j0 + m;
      float dfr = __shfl(dfac, m, 32);
      float pd  = phase_i - sh_phj[buf][m];
      {  // l = ln
        float kv  = kacc[0][r];
        float hjl = sh_hj[buf][m * LL + ln];
        float coh = fast_sig(hca0 + sh_hcBj[buf][m * LL + ln] + bc0);
        float pf  = __cosf(pd * pw0);
        float v   = dfr * kv * aggi + pf * coh * (hjl - hil0);  // -diff = hj-hi
        sum0 += (j == i) ? 0.0f : v;
      }
      {  // l = 16 + ln
        float kv  = kacc[1][r];
        float hjl = sh_hj[buf][m * LL + 16 + ln];
        float coh = fast_sig(hca1 + sh_hcBj[buf][m * LL + 16 + ln] + bc1);
        float pf  = __cosf(pd * pw1);
        float v   = dfr * kv * aggi + pf * coh * (hjl - hil1);
        sum1 += (j == i) ? 0.0f : v;
      }
    }
  }

  // combine lane halves (lane ^ 16 handles the same l, other j rows)
  sum0 += __shfl_xor(sum0, 16, 32);
  sum1 += __shfl_xor(sum1, 16, 32);
  if (lane < 16) {
    part[((size_t)jc * NN + i) * LL + ln]      = sum0;
    part[((size_t)jc * NN + i) * LL + 16 + ln] = sum1;
  }
}

// ---------------------------------------------------------------------------
// Kernel C: reduce partials, write output. 8 waves per block, wave = one i.
// ---------------------------------------------------------------------------
__global__ __launch_bounds__(256) void finalize_kernel(
    const float* __restrict__ self_t, const float* __restrict__ part,
    float* __restrict__ out) {
  int wave = threadIdx.x >> 5;
  int lane = threadIdx.x & 31;
  int i = blockIdx.x * 8 + wave;

  float s = 0.0f;
#pragma unroll
  for (int c = 0; c < JC; ++c) s += part[((size_t)c * NN + i) * LL + lane];

  out[i * (LL + 1) + lane] = 0.5f * self_t[i * LL + lane] + 0.3f * s;

  float ab = fabsf(s);
#pragma unroll
  for (int off = 16; off >= 1; off >>= 1) ab += __shfl_xor(ab, off, 32);
  if (lane == 0) out[i * (LL + 1) + LL] = 0.1f + 0.05f * ab;
}

// ---------------------------------------------------------------------------
extern "C" void kernel_launch(void* const* d_in, const int* in_sizes, int n_in,
                              void* d_out, int out_size, void* d_ws, size_t ws_size,
                              hipStream_t stream) {
  const float* state   = (const float*)d_in[0];
  const float* Ws1     = (const float*)d_in[1];
  const float* bs1     = (const float*)d_in[2];
  const float* Ws2     = (const float*)d_in[3];
  const float* bs2     = (const float*)d_in[4];
  const float* Ws3     = (const float*)d_in[5];
  const float* bs3     = (const float*)d_in[6];
  const float* Wi1     = (const float*)d_in[7];
  const float* bi1     = (const float*)d_in[8];
  const float* Wi2     = (const float*)d_in[9];
  const float* bi2     = (const float*)d_in[10];
  const float* Wi3     = (const float*)d_in[11];
  const float* bi3     = (const float*)d_in[12];
  const float* phase_w = (const float*)d_in[13];
  const float* Wc      = (const float*)d_in[14];
  const float* bc      = (const float*)d_in[15];
  const float* aggr    = (const float*)d_in[16];
  float* out = (float*)d_out;

  float* ws     = (float*)d_ws;
  float* self_t = ws;                      // N*32
  float* av     = self_t + NN * LL;        // N*64
  float* bv     = av + NN * HH;            // N*64
  float* hcA    = bv + NN * HH;            // N*32
  float* hcB    = hcA + NN * LL;           // N*32
  float* aggS   = hcB + NN * LL;           // N
  float* part   = aggS + NN;               // JC*N*32

  node_kernel<<<NN, 64, 0, stream>>>(state, Ws1, bs1, Ws2, bs2, Ws3, bs3,
                                     Wi1, bi1, Wc, aggr,
                                     self_t, av, bv, hcA, hcB, aggS);

  dim3 grid(NN / IWAVES, JC);
  pair_kernel<<<grid, IWAVES * 32, 0, stream>>>(state, Wi2, bi2, Wi3, bi3,
                                                phase_w, bc, av, bv, hcA, hcB,
                                                aggS, part);

  finalize_kernel<<<NN / 8, 256, 0, stream>>>(self_t, part, out);
}
